// SparseAttention_23295902614242
// MI455X (gfx1250) — matrix operand order repaired and checked by reading, not compiled, not run on hardware
//
#include <hip/hip_runtime.h>

// Problem constants (match reference)
#define Bsz   2
#define Sseq  2048
#define DMdl  1024
#define Hn    16
#define Kn    32
#define Dh    64
#define Mrows (Bsz * Sseq)   // 4096

// GEMM tiling
#define BM  128
#define BN  64
#define BK  32
#define LDA 5   // padded row stride in uint4 (4 used + 1 pad) to avoid LDS bank conflicts

typedef __attribute__((ext_vector_type(16))) __bf16 bf16x16;
typedef __attribute__((ext_vector_type(8)))  float  f32x8;
typedef int v4i __attribute__((vector_size(16)));

union FragU { uint4 u4[2]; bf16x16 v; };

__device__ __forceinline__ float bf2f(unsigned int lo16) {
    return __uint_as_float(lo16 << 16);
}
__device__ __forceinline__ unsigned short f2bf(float f) {
    unsigned int u = __float_as_uint(f);
    u += 0x7FFFu + ((u >> 16) & 1u);   // round-to-nearest-even
    return (unsigned short)(u >> 16);
}

// ---------------------------------------------------------------------------
// Async global->LDS (CDNA5 GLOBAL_LOAD_ASYNC_TO_LDS_B128, ASYNCcnt-tracked).
// Signature (from compiler diagnostic): (v4i32 AS1*, v4i32 AS3*, imm, imm).
// Guarded: falls back to a synchronous copy if the builtin is unavailable
// (e.g. on the host-side semantic pass).
// ---------------------------------------------------------------------------
#if defined(__AMDGCN__) && __has_builtin(__builtin_amdgcn_global_load_async_to_lds_b128)
#define USE_ASYNC_LDS 1
#endif

#if defined(USE_ASYNC_LDS)
typedef __attribute__((address_space(1))) v4i* gptr_v4i;
typedef __attribute__((address_space(3))) v4i* lptr_v4i;
#endif

__device__ __forceinline__ void copy16_g2l(const uint4* g, uint4* l) {
#if defined(USE_ASYNC_LDS)
    __builtin_amdgcn_global_load_async_to_lds_b128(
        (gptr_v4i)(uintptr_t)g,
        (lptr_v4i)(unsigned int)(uintptr_t)l,
        0, 0);
#else
    *l = *g;
#endif
}

__device__ __forceinline__ void wait_async_zero() {
#if defined(USE_ASYNC_LDS)
#if __has_builtin(__builtin_amdgcn_s_wait_asynccnt)
    __builtin_amdgcn_s_wait_asynccnt(0);
#else
    asm volatile("s_wait_asynccnt 0x0" ::: "memory");
#endif
#endif
}

// ---------------------------------------------------------------------------
// f32 -> bf16 elementwise conversion
// ---------------------------------------------------------------------------
__global__ void f32_to_bf16_kernel(const float* __restrict__ in,
                                   unsigned short* __restrict__ out, int n) {
    int i = blockIdx.x * blockDim.x + threadIdx.x;
    if (i < n) out[i] = f2bf(in[i]);
}

// ---------------------------------------------------------------------------
// C = A(bf16, MxK row-major) @ W(bf16, NxK row-major)^T  using WMMA bf16.
// Block tile 128x64, BK=32, double-buffered LDS with async global->LDS DMA.
// 8 waves, each owns a 32x32 sub-tile (2x2 of 16x16 WMMA accumulators).
// Output: either bf16 (Cb != null) or f32 (Cf).
// ---------------------------------------------------------------------------
__global__ __launch_bounds__(256)
void gemm_bf16_wmma(const unsigned short* __restrict__ A,
                    const unsigned short* __restrict__ Bw,
                    float* __restrict__ Cf,
                    unsigned short* __restrict__ Cb,
                    int M, int N, int Kd) {
    __shared__ uint4 As[2][BM * LDA];
    __shared__ uint4 Bs[2][BN * LDA];

    const int tid  = threadIdx.x;
    const int lane = tid & 31;
    const int wave = tid >> 5;
    const int m0   = blockIdx.y * BM;
    const int n0   = blockIdx.x * BN;
    const int mSub = (wave >> 1) * 32;   // 4 waves down M
    const int nSub = (wave & 1) * 32;    // 2 waves across N
    const int half = lane >> 4;          // 0: lanes 0-15, 1: lanes 16-31
    const int r    = lane & 15;

    const uint4* Ag = (const uint4*)A;
    const uint4* Bg = (const uint4*)Bw;

    const int rowA0 = tid >> 2, cA = tid & 3;   // A tile: 128 rows x 4 uint4, 2 per thread
    const int rowA1 = rowA0 + 64;
    const int rowB  = tid >> 2, cB = tid & 3;   // B tile: 64 rows x 4 uint4, 1 per thread

    // Per-k-step tile fill: 3 x 16B per thread, direct to LDS (async on CDNA5).
    auto issue_tile = [&](int k0, int buf) {
        copy16_g2l(&Ag[(((size_t)(m0 + rowA0) * Kd + k0) >> 3) + cA], &As[buf][rowA0 * LDA + cA]);
        copy16_g2l(&Ag[(((size_t)(m0 + rowA1) * Kd + k0) >> 3) + cA], &As[buf][rowA1 * LDA + cA]);
        copy16_g2l(&Bg[(((size_t)(n0 + rowB ) * Kd + k0) >> 3) + cB], &Bs[buf][rowB  * LDA + cB]);
    };

    f32x8 c00 = {0.f,0.f,0.f,0.f,0.f,0.f,0.f,0.f};
    f32x8 c01 = c00, c10 = c00, c11 = c00;

    const int nk = Kd / BK;
    issue_tile(0, 0);

    for (int kk = 0; kk < nk; ++kk) {
        const int buf = kk & 1;
        wait_async_zero();      // this wave's tile-kk bytes are in LDS
        __syncthreads();        // everyone's tile-kk in LDS; everyone done reading buf^1
        if (kk + 1 < nk) issue_tile((kk + 1) * BK, buf ^ 1);

        // A fragment (16x32 bf16): lanes 0-15 row M=r hold K{0..7,16..23},
        // lanes 16-31 hold K{8..15,24..31}  -> two b128 LDS reads per fragment.
        FragU a0, a1, b0, b1;
        const uint4* ar0 = &As[buf][(mSub +      r) * LDA];
        const uint4* ar1 = &As[buf][(mSub + 16 + r) * LDA];
        a0.u4[0] = ar0[half];      a0.u4[1] = ar0[2 + half];
        a1.u4[0] = ar1[half];      a1.u4[1] = ar1[2 + half];
        // B fragment (32x16 bf16): lane n holds column n; lanes 0-15 K{0..7,16..23},
        // lanes 16-31 K{8..15,24..31}. Column n of B == row n of W (contiguous).
        const uint4* br0 = &Bs[buf][(nSub +      r) * LDA];
        const uint4* br1 = &Bs[buf][(nSub + 16 + r) * LDA];
        b0.u4[0] = br0[half];      b0.u4[1] = br0[2 + half];
        b1.u4[0] = br1[half];      b1.u4[1] = br1[2 + half];

        c00 = __builtin_amdgcn_wmma_f32_16x16x32_bf16(false, a0.v, false, b0.v, (short)0, c00, false, false);
        c01 = __builtin_amdgcn_wmma_f32_16x16x32_bf16(false, a0.v, false, b1.v, (short)0, c01, false, false);
        c10 = __builtin_amdgcn_wmma_f32_16x16x32_bf16(false, a1.v, false, b0.v, (short)0, c10, false, false);
        c11 = __builtin_amdgcn_wmma_f32_16x16x32_bf16(false, a1.v, false, b1.v, (short)0, c11, false, false);
    }

    // C/D layout: VGPR rr -> row rr + 8*half, col = lane&15
    const int col0 = n0 + nSub + r;
#pragma unroll
    for (int rr = 0; rr < 8; ++rr) {
        const int row0 = m0 + mSub + rr + 8 * half;
        const int row1 = row0 + 16;
        if (Cb) {
            Cb[(size_t)row0 * N + col0     ] = f2bf(c00[rr]);
            Cb[(size_t)row0 * N + col0 + 16] = f2bf(c01[rr]);
            Cb[(size_t)row1 * N + col0     ] = f2bf(c10[rr]);
            Cb[(size_t)row1 * N + col0 + 16] = f2bf(c11[rr]);
        } else {
            Cf[(size_t)row0 * N + col0     ] = c00[rr];
            Cf[(size_t)row0 * N + col0 + 16] = c01[rr];
            Cf[(size_t)row1 * N + col0     ] = c10[rr];
            Cf[(size_t)row1 * N + col0 + 16] = c11[rr];
        }
    }
}

// ---------------------------------------------------------------------------
// Sparse attention: one wave32 per query (b, s, h).
// Phase 1: lane = key j (K=32): score_j = q . K[idx_j] / sqrt(D)
// Softmax across the wave via shuffles.
// Phase 2: lane = dim pair (D=64 -> 2 per lane): out = sum_j p_j * V[idx_j]
// (V reads fully coalesced: 32 lanes read consecutive dwords of one row)
// ---------------------------------------------------------------------------
__global__ __launch_bounds__(256)
void sparse_attn_kernel(const unsigned short* __restrict__ Qb,
                        const unsigned short* __restrict__ Kb,
                        const unsigned short* __restrict__ Vb,
                        const int* __restrict__ idx,
                        const unsigned char* __restrict__ mask,
                        unsigned short* __restrict__ Ob) {
    const int gid  = blockIdx.x * 8 + (threadIdx.x >> 5);  // query id in [0, B*S*H)
    const int lane = threadIdx.x & 31;
    const int h  = gid % Hn;
    const int bs = gid / Hn;            // b*S + s
    const int b  = bs / Sseq;

    const unsigned short* qrow = Qb + (size_t)bs * DMdl + h * Dh;

    const int           ij = idx [(size_t)bs * Kn + lane];
    const unsigned char mj = mask[(size_t)bs * Kn + lane];
    const unsigned short* krow = Kb + ((size_t)(b * Sseq + ij)) * DMdl + h * Dh;

    // score for this lane's key
    const uint2* q2 = (const uint2*)qrow;
    const uint2* k2 = (const uint2*)krow;
    float score = 0.f;
#pragma unroll
    for (int cc = 0; cc < 16; ++cc) {
        uint2 qa = q2[cc], ka = k2[cc];
        score += bf2f(qa.x & 0xFFFFu) * bf2f(ka.x & 0xFFFFu);
        score += bf2f(qa.x >> 16)     * bf2f(ka.x >> 16);
        score += bf2f(qa.y & 0xFFFFu) * bf2f(ka.y & 0xFFFFu);
        score += bf2f(qa.y >> 16)     * bf2f(ka.y >> 16);
    }
    score *= 0.125f;                     // 1/sqrt(64)
    if (!mj) score = -INFINITY;

    // wave32 softmax
    float mx = score;
#pragma unroll
    for (int o = 16; o > 0; o >>= 1) mx = fmaxf(mx, __shfl_xor(mx, o, 32));
    float e = (mx == -INFINITY) ? 0.f : __expf(score - mx);
    float ssum = e;
#pragma unroll
    for (int o = 16; o > 0; o >>= 1) ssum += __shfl_xor(ssum, o, 32);
    const float p = e / (ssum > 0.f ? ssum : 1.f);

    // weighted V accumulation; lane owns dims {2*lane, 2*lane+1}
    float a0 = 0.f, a1 = 0.f;
#pragma unroll 4
    for (int t = 0; t < Kn; ++t) {
        const float pt = __shfl(p, t, 32);
        const int   it = __shfl(ij, t, 32);
        const unsigned int vv =
            ((const unsigned int*)(Vb + ((size_t)(b * Sseq + it)) * DMdl + h * Dh))[lane];
        a0 = fmaf(pt, bf2f(vv & 0xFFFFu), a0);
        a1 = fmaf(pt, bf2f(vv >> 16),     a1);
    }
    const unsigned int packed = (unsigned int)f2bf(a0) | ((unsigned int)f2bf(a1) << 16);
    ((unsigned int*)(Ob + (size_t)bs * DMdl + h * Dh))[lane] = packed;
}

// ---------------------------------------------------------------------------
// Host launcher
// ---------------------------------------------------------------------------
extern "C" void kernel_launch(void* const* d_in, const int* in_sizes, int n_in,
                              void* d_out, int out_size, void* d_ws, size_t ws_size,
                              hipStream_t stream) {
    (void)in_sizes; (void)n_in; (void)out_size; (void)ws_size;

    const float*         x    = (const float*)d_in[0];
    const int*           aidx = (const int*)d_in[1];
    const unsigned char* amsk = (const unsigned char*)d_in[2];
    const float*         Wq   = (const float*)d_in[3];
    const float*         Wk   = (const float*)d_in[4];
    const float*         Wv   = (const float*)d_in[5];
    const float*         Wo   = (const float*)d_in[6];
    float*               out  = (float*)d_out;

    const size_t NX = (size_t)Mrows * DMdl;   // 4M elems
    const size_t NW = (size_t)DMdl * DMdl;    // 1M elems

    unsigned short* w = (unsigned short*)d_ws;
    unsigned short* xb    = w;                 w += NX;
    unsigned short* wqb   = w;                 w += NW;
    unsigned short* wkb   = w;                 w += NW;
    unsigned short* wvb   = w;                 w += NW;
    unsigned short* wob   = w;                 w += NW;
    unsigned short* Qb    = w;                 w += NX;
    unsigned short* Kbuf  = w;                 w += NX;
    unsigned short* Vbuf  = w;                 w += NX;
    unsigned short* attnb = w;                 w += NX;   // total ~48 MiB

    // 1) convert inputs to bf16
    f32_to_bf16_kernel<<<(unsigned)((NX + 255) / 256), 256, 0, stream>>>(x,  xb,  (int)NX);
    f32_to_bf16_kernel<<<(unsigned)((NW + 255) / 256), 256, 0, stream>>>(Wq, wqb, (int)NW);
    f32_to_bf16_kernel<<<(unsigned)((NW + 255) / 256), 256, 0, stream>>>(Wk, wkb, (int)NW);
    f32_to_bf16_kernel<<<(unsigned)((NW + 255) / 256), 256, 0, stream>>>(Wv, wvb, (int)NW);
    f32_to_bf16_kernel<<<(unsigned)((NW + 255) / 256), 256, 0, stream>>>(Wo, wob, (int)NW);

    // 2) Q/K/V projections (y = x @ W^T), bf16 outputs
    dim3 gg(DMdl / BN, Mrows / BM);   // (16, 32)
    gemm_bf16_wmma<<<gg, 256, 0, stream>>>(xb, wqb, nullptr, Qb,   Mrows, DMdl, DMdl);
    gemm_bf16_wmma<<<gg, 256, 0, stream>>>(xb, wkb, nullptr, Kbuf, Mrows, DMdl, DMdl);
    gemm_bf16_wmma<<<gg, 256, 0, stream>>>(xb, wvb, nullptr, Vbuf, Mrows, DMdl, DMdl);

    // 3) gather + softmax + PV
    sparse_attn_kernel<<<(Bsz * Sseq * Hn) / 8, 256, 0, stream>>>(Qb, Kbuf, Vbuf, aidx, amsk, attnb);

    // 4) output projection, f32 into d_out
    gemm_bf16_wmma<<<gg, 256, 0, stream>>>(attnb, wob, out, nullptr, Mrows, DMdl, DMdl);
}
